// Conv2dFusion_65704409694920
// MI455X (gfx1250) — matrix operand order repaired and checked by reading, not compiled, not run on hardware
//
#include <hip/hip_runtime.h>

typedef _Float16 half8  __attribute__((ext_vector_type(8)));
typedef _Float16 half16 __attribute__((ext_vector_type(16)));
typedef float    float8 __attribute__((ext_vector_type(8)));

#define NIMG 512      // B*S
#define D_   256
#define H1   127      // conv1 out (stride 2, VALID)
#define P1   42       // after pool3
#define H2   40       // conv2 out
#define P2   13
#define H3   11       // conv3 out
#define NEG_HUGE (-3.402823466e+38f)

// Branch-free GELU (tanh form rewritten as sigmoid): x * sigma(1.5957691x + 0.0713548x^3)
// Uses hardware v_exp_f32 + v_rcp_f32; error ~3e-4, below the f16 activation ULP.
__device__ __forceinline__ float gelu_fast(float x) {
    float z = x * (1.59576912f + 0.07135481f * x * x);
    float e = __expf(-z);                       // v_exp_f32 path
    return x * __builtin_amdgcn_rcpf(1.0f + e); // v_rcp_f32
}

// ---------------------------------------------------------------------------
// Kernel 1: rank-1 conv1 + GELU + maxpool3 -> x1 (f16, NHWC [img][42][42][16]).
// fused[i,j] = u[i]*v[j]  =>  conv1(c,oy,ox) = b1[c] + sum_ky u[2oy+ky]*h[c,ky,ox]
// h[c,ky,ox] = sum_kx w1[c,0,ky,kx]*v[2ox+kx].  (Never materializes the 134MB
// outer-product tensor.)  NHWC layout makes conv2's WMMA A-fragments
// contiguous 16-byte chunks.
// ---------------------------------------------------------------------------
__global__ __launch_bounds__(256) void conv1_kernel(
    const float* __restrict__ token,   // v (cols)
    const float* __restrict__ type_e,  // u (rows)
    const float* __restrict__ w1, const float* __restrict__ b1,
    _Float16* __restrict__ x1)         // [512][42][42][16]
{
    __shared__ float u[D_];
    __shared__ float v[D_];
    __shared__ float h[16 * 3 * H1];

    const int img = blockIdx.x;
    const int tid = threadIdx.x;

    u[tid] = type_e[img * D_ + tid];
    v[tid] = token[img * D_ + tid];
    __syncthreads();

    for (int idx = tid; idx < 16 * 3 * H1; idx += 256) {
        int c  = idx / (3 * H1);
        int r  = idx % (3 * H1);
        int ky = r / H1;
        int ox = r % H1;
        const float* wr = w1 + (c * 3 + ky) * 3;
        h[idx] = wr[0] * v[2 * ox] + wr[1] * v[2 * ox + 1] + wr[2] * v[2 * ox + 2];
    }
    __syncthreads();

    // idx = ((py*42+px)*16 + c): channel-minor => coalesced NHWC stores
    for (int idx = tid; idx < 16 * P1 * P1; idx += 256) {
        int c  = idx & 15;
        int r  = idx >> 4;           // py*42+px
        int py = r / P1, px = r % P1;
        float bias = b1[c];
        float mx = NEG_HUGE;
        for (int dy = 0; dy < 3; ++dy) {
            int oy = 3 * py + dy;
            float u0 = u[2 * oy], u1 = u[2 * oy + 1], u2 = u[2 * oy + 2];
            for (int dx = 0; dx < 3; ++dx) {
                int ox = 3 * px + dx;
                const float* hp = h + c * 3 * H1 + ox;
                float conv = bias + u0 * hp[0] + u1 * hp[H1] + u2 * hp[2 * H1];
                mx = fmaxf(mx, gelu_fast(conv));
            }
        }
        x1[(size_t)img * (P1 * P1 * 16) + idx] = (_Float16)mx;
    }
}

// ---------------------------------------------------------------------------
// Kernel 2: conv2 + conv3 as implicit GEMMs on v_wmma_f32_16x16x32_f16.
//   conv2: M=1600 (40x40), N=32, K=144 perm k=(ky*3+kx)*16+ci  (pad 160)
//   conv3: M=121 (pad 128), N=32, K=288 perm k=(ky*3+kx)*32+ci
// NHWC activations => every lane's A-fragment = two contiguous b128 loads
// (global/L2 for conv2, LDS for conv3). B fragments hoisted into registers
// once per wave. D tiles written back as single b128 vector stores.
// Fragment layouts per ISA 7.12.2 (wave32):
//   A 16x32 f16 : lane l -> M=l%16, hi=l>>4 ; elem j -> K = 8*hi + j + (j>=8?8:0)
//   B 32x16 f16 : lane l -> N=l%16          ; elem j -> K = 8*hi + j + (j>=8?8:0)
//   D 16x16 f32 : elem r -> M = r + 8*hi    ; N = l%16
// ---------------------------------------------------------------------------
__global__ __launch_bounds__(256) void conv23_kernel(
    const _Float16* __restrict__ x1,   // [512][42][42][16] NHWC
    const float* __restrict__ w2, const float* __restrict__ b2,
    const float* __restrict__ w3, const float* __restrict__ b3,
    float* __restrict__ out)           // [512][288]
{
    __shared__ __align__(16) _Float16 buf2[16 * H2 * H2]; // 51.2KB; aliased as f32 buf3
    __shared__ __align__(16) _Float16 x2s[P2 * P2 * 32];  // NHWC, 10.8KB
    float* buf3 = reinterpret_cast<float*>(buf2);         // [32][128] f32 = 16KB (padded)

    const int img  = blockIdx.x;
    const int tid  = threadIdx.x;
    const int lane = tid & 31;
    const int wave = tid >> 5;
    const int hi   = lane >> 4;
    const int ln16 = lane & 15;

    const _Float16* x1g = x1 + (size_t)img * (P1 * P1 * 16);
    const _Float16* ap  = x1g + 8 * hi;   // lane's channel-chunk base

    // ================= conv2 (16->32), two 16-channel halves =================
    for (int nt = 0; nt < 2; ++nt) {
        const int n = nt * 16 + ln16;
        // hoisted B fragments: 5 steps, gathered once, register-resident
        half16 B2[5];
        #pragma unroll
        for (int kk = 0; kk < 5; ++kk) {
            half16 bm;
            #pragma unroll
            for (int j = 0; j < 16; ++j) {
                int kb = kk * 32 + 8 * hi + j + (j >= 8 ? 8 : 0);
                _Float16 bv = (_Float16)0.f;
                if (kb < 144) {
                    int s = kb >> 4, ci = kb & 15;          // k = s*16 + ci
                    bv = (_Float16)w2[n * 144 + ci * 9 + s];
                }
                bm[j] = bv;
            }
            B2[kk] = bm;
        }

        for (int mt = wave; mt < 100; mt += 8) {   // wave-uniform: EXEC stays full
            const int m  = mt * 16 + ln16;
            const int oy = m / H2, ox = m % H2;
            if (mt + 8 < 100) {                    // prefetch next tile's A base
                int m2 = (mt + 8) * 16 + ln16;
                __builtin_prefetch(ap + ((m2 / H2) * P1 + (m2 % H2)) * 16, 0, 1);
            }
            float8 acc = {0.f, 0.f, 0.f, 0.f, 0.f, 0.f, 0.f, 0.f};
            #pragma unroll
            for (int kk = 0; kk < 5; ++kk) {
                const int s0 = 2 * kk;
                const int ky0 = s0 / 3, kx0 = s0 % 3;
                half8 c0 = *(const half8*)(ap + ((oy + ky0) * P1 + (ox + kx0)) * 16);
                half8 c1 = {};
                if (kk < 4) {
                    const int s1 = s0 + 1;
                    const int ky1 = s1 / 3, kx1 = s1 % 3;
                    c1 = *(const half8*)(ap + ((oy + ky1) * P1 + (ox + kx1)) * 16);
                }
                half16 a = __builtin_shufflevector(c0, c1,
                            0,1,2,3,4,5,6,7,8,9,10,11,12,13,14,15);
                acc = __builtin_amdgcn_wmma_f32_16x16x32_f16(
                          false, a, false, B2[kk], (short)0, acc, false, false);
            }
            const float bias = b2[n];
            half8 o;                               // 8 consecutive M => one b128 store
            #pragma unroll
            for (int r = 0; r < 8; ++r)
                o[r] = (_Float16)gelu_fast(acc[r] + bias);
            *reinterpret_cast<half8*>(buf2 + ln16 * (H2 * H2) + mt * 16 + 8 * hi) = o;
        }
        __syncthreads();
        // maxpool3 -> x2 NHWC channels [nt*16, nt*16+16)
        for (int idx = tid; idx < 16 * P2 * P2; idx += 256) {
            int c  = idx & 15;
            int r  = idx >> 4;                     // py*13+px
            int py = r / P2, px = r % P2;
            float mx = NEG_HUGE;
            for (int dy = 0; dy < 3; ++dy)
                for (int dx = 0; dx < 3; ++dx)
                    mx = fmaxf(mx, (float)buf2[c * (H2 * H2) + (3 * py + dy) * H2 + (3 * px + dx)]);
            x2s[r * 32 + nt * 16 + c] = (_Float16)mx;
        }
        __syncthreads();
    }

    // ================= conv3 (32->32), M=121 padded to 128 ===================
    {
        const int nt = wave & 1;                   // each wave owns a fixed n-half
        const int n  = nt * 16 + ln16;
        half16 B3[9];
        #pragma unroll
        for (int kk = 0; kk < 9; ++kk) {
            half16 bm;
            #pragma unroll
            for (int j = 0; j < 16; ++j) {
                int ci = 8 * hi + j + (j >= 8 ? 8 : 0); // k = kk*32 + ci
                bm[j] = (_Float16)w3[n * 288 + ci * 9 + kk];
            }
            B3[kk] = bm;
        }
        const float bias = b3[n];
        for (int t = wave; t < 16; t += 8) {       // mt = wave>>1 and wave>>1+4
            const int mt = t >> 1;
            float8 acc = {0.f, 0.f, 0.f, 0.f, 0.f, 0.f, 0.f, 0.f};
            const int m  = mt * 16 + ln16;
            const int mc = (m < 121) ? m : 0;
            const int oy = mc / H3, ox = mc % H3;
            #pragma unroll
            for (int kk = 0; kk < 9; ++kk) {
                const int ky = kk / 3, kx = kk % 3;
                const _Float16* rp = x2s + ((oy + ky) * P2 + (ox + kx)) * 32;
                half8 c0 = *(const half8*)(rp + 8 * hi);        // ci 8hi..8hi+7
                half8 c1 = *(const half8*)(rp + 16 + 8 * hi);   // ci 16+8hi..
                half16 a = __builtin_shufflevector(c0, c1,
                            0,1,2,3,4,5,6,7,8,9,10,11,12,13,14,15);
                acc = __builtin_amdgcn_wmma_f32_16x16x32_f16(
                          false, a, false, B3[kk], (short)0, acc, false, false);
            }
            float8 o;                              // padded stride 128: no guard needed
            #pragma unroll
            for (int r = 0; r < 8; ++r)
                o[r] = gelu_fast(acc[r] + bias);
            *reinterpret_cast<float8*>(buf3 + n * 128 + mt * 16 + 8 * hi) = o;
        }
    }
    __syncthreads();

    // final maxpool3 -> out[img][c*9 + py*3 + px]
    for (int idx = tid; idx < 32 * 9; idx += 256) {
        int c  = idx / 9;
        int r  = idx % 9;
        int py = r / 3, px = r % 3;
        float mx = NEG_HUGE;
        for (int dy = 0; dy < 3; ++dy)
            for (int dx = 0; dx < 3; ++dx)
                mx = fmaxf(mx, buf3[c * 128 + (3 * py + dy) * H3 + (3 * px + dx)]);
        out[(size_t)img * 288 + idx] = mx;
    }
}

extern "C" void kernel_launch(void* const* d_in, const int* in_sizes, int n_in,
                              void* d_out, int out_size, void* d_ws, size_t ws_size,
                              hipStream_t stream) {
    const float* token  = (const float*)d_in[0];
    const float* type_e = (const float*)d_in[1];
    const float* w1 = (const float*)d_in[2];
    const float* b1 = (const float*)d_in[3];
    const float* w2 = (const float*)d_in[4];
    const float* b2 = (const float*)d_in[5];
    const float* w3 = (const float*)d_in[6];
    const float* b3 = (const float*)d_in[7];
    float* out = (float*)d_out;

    _Float16* x1 = (_Float16*)d_ws;   // 512*42*42*16 halves = 28.9 MB scratch

    conv1_kernel<<<NIMG, 256, 0, stream>>>(token, type_e, w1, b1, x1);
    conv23_kernel<<<NIMG, 256, 0, stream>>>(x1, w2, b2, w3, b3, out);
}
